// Gated_78408922956183
// MI455X (gfx1250) — compile-verified, matrix operand order checked by reading, not run on hardware
//
#include <hip/hip_runtime.h>
#include <cstdint>

#define NN 65536
#define DD 256
#define GFD 64
#define EE 262144

typedef __attribute__((ext_vector_type(16))) __bf16 v16bf;
typedef __attribute__((ext_vector_type(8)))  float  v8f;

union FragBF { v16bf v; unsigned u[8]; };

// hardware f32 -> bf16 (RTNE) via native convert
__device__ __forceinline__ unsigned short f2bf(float f) {
    union { __bf16 b; unsigned short s; } u;
    u.b = (__bf16)f;
    return u.s;
}

__device__ __forceinline__ float sigm(float x) { return 1.0f / (1.0f + expf(-x)); }

// Load a 16x32 bf16 WMMA fragment from an LDS row of >=32 bf16 (dword-aligned).
// Per CDNA5 ISA layout: lanes 0-15 hold K pairs {0..3},{8..11}; lanes 16-31 get +4.
__device__ __forceinline__ void load_frag(FragBF& f, const void* rowPtr, int lane) {
    const unsigned* p = (const unsigned*)rowPtr;
    int pb = (lane >> 4) ? 4 : 0;
#pragma unroll
    for (int d = 0; d < 8; ++d) {
        int pair = (d < 4 ? d : d + 4) + pb;
        f.u[d] = p[pair];
    }
}

// ---------------------------------------------------------------------------
// Weight packing: f32 [K][N] -> bf16 [N][K]   (transpose + convert)
__global__ void packT_kernel(const float* __restrict__ src, unsigned short* __restrict__ dst,
                             int K, int N) {
    int idx = blockIdx.x * 256 + threadIdx.x;
    if (idx < K * N) {
        int n = idx / K, k = idx - n * K;
        dst[idx] = f2bf(src[(size_t)k * N + n]);
    }
}

// f32 -> bf16 elementwise (layout already [N][K])
__global__ void pack_kernel(const float* __restrict__ src, unsigned short* __restrict__ dst,
                            int count) {
    int idx = blockIdx.x * 256 + threadIdx.x;
    if (idx < count) dst[idx] = f2bf(src[idx]);
}

__global__ void zero_kernel(float* __restrict__ p, int count) {
    int idx = blockIdx.x * 256 + threadIdx.x;
    if (idx < count) p[idx] = 0.0f;
}

// ---------------------------------------------------------------------------
// Generic GEMM: C[M,Ntot] = act(A[M,K] @ B + bias), B pre-packed bf16 [Ntot][K].
// Block = 256 threads (8 waves). Tile: 128 rows x 128 cols; each wave covers
// 16 rows x 128 cols = 8 WMMA per K-step (A fragment amortized over 8 B tiles).
__global__ void __launch_bounds__(256)
gemm_kernel(const float* __restrict__ A, const unsigned short* __restrict__ Bp,
            const float* __restrict__ bias, float* __restrict__ C,
            int Ntot, int K, int act, int reluA) {
    __shared__ __bf16          sA[128][34];
    __shared__ unsigned short  sB[128][34];

    int tid = threadIdx.x;
    int wave = tid >> 5, lane = tid & 31;
    int rowBase = blockIdx.x * 128;
    int colBase = blockIdx.y * 128;

    v8f acc[8] = {};

    for (int k0 = 0; k0 < K; k0 += 32) {
        {   // stage A tile (f32 -> bf16), 16 contiguous floats per thread
            int r = tid >> 1, hf = (tid & 1) * 16;
            const float* src = A + (size_t)(rowBase + r) * K + k0 + hf;
            __bf16* dst = &sA[r][hf];
#pragma unroll
            for (int i = 0; i < 16; ++i) {
                float fv = src[i];
                if (reluA) fv = fmaxf(fv, 0.0f);
                dst[i] = (__bf16)fv;
            }
        }
        {   // stage B tile: 16 contiguous bf16 per thread
            int c = tid >> 1, hf = (tid & 1) * 16;
            const unsigned short* src = Bp + (size_t)(colBase + c) * K + k0 + hf;
            unsigned short* dst = &sB[c][hf];
#pragma unroll
            for (int i = 0; i < 16; ++i) dst[i] = src[i];
        }
        __syncthreads();

        FragBF fa;
        load_frag(fa, &sA[wave * 16 + (lane & 15)][0], lane);
#pragma unroll
        for (int nt = 0; nt < 8; ++nt) {
            FragBF fb;
            load_frag(fb, &sB[nt * 16 + (lane & 15)][0], lane);
            acc[nt] = __builtin_amdgcn_wmma_f32_16x16x32_bf16(
                false, fa.v, false, fb.v, (short)0, acc[nt], false, false);
        }
        __syncthreads();
    }

    int rbase = rowBase + wave * 16 + ((lane >> 4) ? 8 : 0);
#pragma unroll
    for (int nt = 0; nt < 8; ++nt) {
        int c = colBase + nt * 16 + (lane & 15);
        float bv = bias ? bias[c] : 0.0f;
#pragma unroll
        for (int e = 0; e < 8; ++e) {
            float v = acc[nt][e] + bv;
            if (act) v = fmaxf(v, 0.0f);
            C[(size_t)(rbase + e) * Ntot + c] = v;
        }
    }
}

// ---------------------------------------------------------------------------
// Fused GRU cell: per wave, 16 rows x 32 cols of all six gate pre-activations
// (12 accumulators, 12 WMMA per K-step), gating epilogue in registers.
__global__ void __launch_bounds__(256)
gru_kernel(const float* __restrict__ magg, const float* __restrict__ hin,
           const unsigned short* __restrict__ WihP, const unsigned short* __restrict__ WhhP,
           const float* __restrict__ bih, const float* __restrict__ bhh,
           float* __restrict__ hout) {
    __shared__ __bf16         sM[128][34];
    __shared__ __bf16         sH[128][34];
    __shared__ unsigned short sBi[3][32][34];
    __shared__ unsigned short sBh[3][32][34];

    int tid = threadIdx.x;
    int wave = tid >> 5, lane = tid & 31;
    int rowBase = blockIdx.x * 128;
    int c0 = blockIdx.y * 32;

    v8f aIr[2] = {}, aIz[2] = {}, aIn[2] = {}, aHr[2] = {}, aHz[2] = {}, aHn[2] = {};

    for (int k0 = 0; k0 < DD; k0 += 32) {
        {   // stage m and h tiles
            int r = tid >> 1, hf = (tid & 1) * 16;
            const float* sm = magg + (size_t)(rowBase + r) * DD + k0 + hf;
            const float* sh = hin + (size_t)(rowBase + r) * DD + k0 + hf;
            __bf16* dm = &sM[r][hf];
            __bf16* dh = &sH[r][hf];
#pragma unroll
            for (int i = 0; i < 16; ++i) { dm[i] = (__bf16)sm[i]; dh[i] = (__bf16)sh[i]; }
        }
        if (tid < 192) {  // stage 6 weight tiles of 32 cols x 32 K (32 bf16 per thread)
            int t = tid >> 5;            // 0..5
            int c = tid & 31;            // 0..31
            int g = (t < 3) ? t : t - 3;
            const unsigned short* W = (t < 3) ? WihP : WhhP;
            const unsigned short* src = W + (size_t)(g * DD + c0 + c) * DD + k0;
            unsigned short* dst = (t < 3) ? &sBi[g][c][0] : &sBh[g][c][0];
#pragma unroll
            for (int i = 0; i < 32; ++i) dst[i] = src[i];
        }
        __syncthreads();

        FragBF fm, fh;
        load_frag(fm, &sM[wave * 16 + (lane & 15)][0], lane);
        load_frag(fh, &sH[wave * 16 + (lane & 15)][0], lane);
#pragma unroll
        for (int ct = 0; ct < 2; ++ct) {
            int cc = ct * 16 + (lane & 15);
            FragBF b0, b1, b2, b3, b4, b5;
            load_frag(b0, &sBi[0][cc][0], lane);
            load_frag(b1, &sBi[1][cc][0], lane);
            load_frag(b2, &sBi[2][cc][0], lane);
            load_frag(b3, &sBh[0][cc][0], lane);
            load_frag(b4, &sBh[1][cc][0], lane);
            load_frag(b5, &sBh[2][cc][0], lane);
            aIr[ct] = __builtin_amdgcn_wmma_f32_16x16x32_bf16(false, fm.v, false, b0.v, (short)0, aIr[ct], false, false);
            aIz[ct] = __builtin_amdgcn_wmma_f32_16x16x32_bf16(false, fm.v, false, b1.v, (short)0, aIz[ct], false, false);
            aIn[ct] = __builtin_amdgcn_wmma_f32_16x16x32_bf16(false, fm.v, false, b2.v, (short)0, aIn[ct], false, false);
            aHr[ct] = __builtin_amdgcn_wmma_f32_16x16x32_bf16(false, fh.v, false, b3.v, (short)0, aHr[ct], false, false);
            aHz[ct] = __builtin_amdgcn_wmma_f32_16x16x32_bf16(false, fh.v, false, b4.v, (short)0, aHz[ct], false, false);
            aHn[ct] = __builtin_amdgcn_wmma_f32_16x16x32_bf16(false, fh.v, false, b5.v, (short)0, aHn[ct], false, false);
        }
        __syncthreads();
    }

    int rbase = rowBase + wave * 16 + ((lane >> 4) ? 8 : 0);
#pragma unroll
    for (int ct = 0; ct < 2; ++ct) {
        int col = c0 + ct * 16 + (lane & 15);
        float bir = bih[col], biz = bih[col + DD], bin = bih[col + 2 * DD];
        float bhr = bhh[col], bhz = bhh[col + DD], bhn = bhh[col + 2 * DD];
#pragma unroll
        for (int e = 0; e < 8; ++e) {
            size_t idx = (size_t)(rbase + e) * DD + col;
            float r = sigm(aIr[ct][e] + bir + aHr[ct][e] + bhr);
            float z = sigm(aIz[ct][e] + biz + aHz[ct][e] + bhz);
            float n = tanhf(aIn[ct][e] + bin + r * (aHn[ct][e] + bhn));
            float hp = hin[idx];
            hout[idx] = (1.0f - z) * n + z * hp;
        }
    }
}

// ---------------------------------------------------------------------------
// Edge scatter-add: 4 edges per block, one lane owns 4 consecutive floats.
// float4 read + native FP32 global atomics.
__global__ void scatter_kernel(const float* __restrict__ mpre, const int* __restrict__ srcIdx,
                               const int* __restrict__ dstIdx, float* __restrict__ magg) {
    int e = blockIdx.x * 4 + (threadIdx.x >> 6);
    int c = (threadIdx.x & 63) * 4;
    int s = srcIdx[e], d = dstIdx[e];
    const float4 v = *(const float4*)&mpre[(size_t)s * DD + c];
    float* dp = &magg[(size_t)d * DD + c];
    unsafeAtomicAdd(dp + 0, v.x);
    unsafeAtomicAdd(dp + 1, v.y);
    unsafeAtomicAdd(dp + 2, v.z);
    unsafeAtomicAdd(dp + 3, v.w);
}

// Per-column sum / sum-of-squares for BatchNorm
__global__ void colstats_kernel(const float* __restrict__ h0, float* __restrict__ sums,
                                float* __restrict__ sumsq) {
    int col = threadIdx.x;
    int r0 = blockIdx.x * 256;
    float s = 0.0f, q = 0.0f;
    for (int r = r0; r < r0 + 256; ++r) {
        float v = h0[(size_t)r * DD + col];
        s += v; q += v * v;
    }
    unsafeAtomicAdd(&sums[col], s);
    unsafeAtomicAdd(&sumsq[col], q);
}

__global__ void bn_kernel(const float* __restrict__ h0, const float* __restrict__ sums,
                          const float* __restrict__ sumsq, const float* __restrict__ gamma,
                          const float* __restrict__ beta, float* __restrict__ out) {
    size_t i = (size_t)blockIdx.x * 256 + threadIdx.x;
    int col = (int)(i & (DD - 1));
    float mu = sums[col] * (1.0f / (float)NN);
    float var = sumsq[col] * (1.0f / (float)NN) - mu * mu;
    out[i] = (h0[i] - mu) * rsqrtf(var + 1e-5f) * gamma[col] + beta[col];
}

// y2[i] = dot(y1[i,:], w) + b   (one wave per node)
__global__ void dot_kernel(const float* __restrict__ y1, const float* __restrict__ w,
                           const float* __restrict__ bp, int m, float* __restrict__ y2) {
    int wave = threadIdx.x >> 5, lane = threadIdx.x & 31;
    int node = blockIdx.x * 8 + wave;
    const float* row = y1 + (size_t)node * DD;
    float s = 0.0f;
#pragma unroll
    for (int i = lane; i < DD; i += 32) s += row[i] * w[i];
#pragma unroll
    for (int off = 16; off > 0; off >>= 1) s += __shfl_down(s, off, 32);
    if (lane == 0) y2[node] = s + bp[m];
}

// bootstrap: mean over 4 mixtures, std = sqrt(var(ddof=1) + eps)
__global__ void boot_kernel(const float* __restrict__ y2, float* __restrict__ out) {
    int i = blockIdx.x * 256 + threadIdx.x;
    float a = y2[i], b = y2[NN + i], c = y2[2 * NN + i], d = y2[3 * NN + i];
    float mu = 0.25f * (a + b + c + d);
    float va = ((a - mu) * (a - mu) + (b - mu) * (b - mu) +
                (c - mu) * (c - mu) + (d - mu) * (d - mu)) * (1.0f / 3.0f);
    out[i] = mu;
    out[NN + i] = sqrtf(va + 1e-5f);
}

// ---------------------------------------------------------------------------
extern "C" void kernel_launch(void* const* d_in, const int* in_sizes, int n_in,
                              void* d_out, int out_size, void* d_ws, size_t ws_size,
                              hipStream_t stream) {
    const float* x       = (const float*)d_in[0];
    const int*   edges   = (const int*)d_in[1];
    const float* W_embed = (const float*)d_in[2];
    const float* b_embed = (const float*)d_in[3];
    const float* bn_g    = (const float*)d_in[4];
    const float* bn_b    = (const float*)d_in[5];
    const float* ggc_W   = (const float*)d_in[6];
    const float* gru_Wih = (const float*)d_in[7];
    const float* gru_Whh = (const float*)d_in[8];
    const float* gru_bih = (const float*)d_in[9];
    const float* gru_bhh = (const float*)d_in[10];
    const float* mlp_W0  = (const float*)d_in[11];
    const float* mlp_b0  = (const float*)d_in[12];
    const float* mlp_W1  = (const float*)d_in[13];
    const float* mlp_b1  = (const float*)d_in[14];
    const float* mlp_W2  = (const float*)d_in[15];
    const float* mlp_b2  = (const float*)d_in[16];
    const int* eSrc = edges;
    const int* eDst = edges + EE;

    char* p = (char*)d_ws;
    auto alloc = [&](size_t bytes) -> void* {
        void* r = (void*)p;
        p += (bytes + 255) & ~(size_t)255;
        return r;
    };
    float* hA   = (float*)alloc(sizeof(float) * NN * DD);   // h ping
    float* hB   = (float*)alloc(sizeof(float) * NN * DD);   // h pong
    float* bufC = (float*)alloc(sizeof(float) * NN * DD);   // h0 / mpre / y0
    float* bufD = (float*)alloc(sizeof(float) * NN * DD);   // magg / y1
    float* stats = (float*)alloc(sizeof(float) * 2 * DD);   // sums | sumsq
    float* y2    = (float*)alloc(sizeof(float) * 4 * NN);
    unsigned short* embedT = (unsigned short*)alloc(2u * GFD * DD);
    unsigned short* ggcT   = (unsigned short*)alloc(2u * 4 * DD * DD);
    unsigned short* WihP   = (unsigned short*)alloc(2u * 3 * DD * DD);
    unsigned short* WhhP   = (unsigned short*)alloc(2u * 3 * DD * DD);
    unsigned short* mlp0T  = (unsigned short*)alloc(2u * 4 * DD * DD);
    unsigned short* mlp1T  = (unsigned short*)alloc(2u * 4 * DD * DD);
    float* sums = stats;
    float* sumsq = stats + DD;

    // ---- pack weights to bf16 ----
    packT_kernel<<<(GFD * DD + 255) / 256, 256, 0, stream>>>(W_embed, embedT, GFD, DD);
    for (int l = 0; l < 4; ++l)
        packT_kernel<<<(DD * DD + 255) / 256, 256, 0, stream>>>(ggc_W + (size_t)l * DD * DD,
                                                                ggcT + (size_t)l * DD * DD, DD, DD);
    pack_kernel<<<(3 * DD * DD + 255) / 256, 256, 0, stream>>>(gru_Wih, WihP, 3 * DD * DD);
    pack_kernel<<<(3 * DD * DD + 255) / 256, 256, 0, stream>>>(gru_Whh, WhhP, 3 * DD * DD);
    for (int m = 0; m < 4; ++m) {
        packT_kernel<<<(DD * DD + 255) / 256, 256, 0, stream>>>(mlp_W0 + (size_t)m * DD * DD,
                                                                mlp0T + (size_t)m * DD * DD, DD, DD);
        packT_kernel<<<(DD * DD + 255) / 256, 256, 0, stream>>>(mlp_W1 + (size_t)m * DD * DD,
                                                                mlp1T + (size_t)m * DD * DD, DD, DD);
    }

    // ---- embed + BatchNorm ----
    gemm_kernel<<<dim3(NN / 128, DD / 128), 256, 0, stream>>>(x, embedT, b_embed, bufC,
                                                              DD, GFD, 0, 0);
    zero_kernel<<<2, 256, 0, stream>>>(stats, 2 * DD);
    colstats_kernel<<<NN / 256, 256, 0, stream>>>(bufC, sums, sumsq);
    bn_kernel<<<(NN * DD) / 256, 256, 0, stream>>>(bufC, sums, sumsq, bn_g, bn_b, hA);

    // ---- 4 GatedGraphConv layers ----
    float* hin = hA;
    float* hout = hB;
    for (int l = 0; l < 4; ++l) {
        gemm_kernel<<<dim3(NN / 128, DD / 128), 256, 0, stream>>>(
            hin, ggcT + (size_t)l * DD * DD, nullptr, bufC, DD, DD, 0, 0);
        zero_kernel<<<(NN * DD) / 256, 256, 0, stream>>>(bufD, NN * DD);
        scatter_kernel<<<EE / 4, 256, 0, stream>>>(bufC, eSrc, eDst, bufD);
        gru_kernel<<<dim3(NN / 128, DD / 32), 256, 0, stream>>>(
            bufD, hin, WihP, WhhP, gru_bih, gru_bhh, hout);
        float* t = hin; hin = hout; hout = t;
    }

    // ---- mixture MLPs (ReLU on h fused into A-stage of first GEMM) ----
    for (int m = 0; m < 4; ++m) {
        gemm_kernel<<<dim3(NN / 128, DD / 128), 256, 0, stream>>>(
            hin, mlp0T + (size_t)m * DD * DD, mlp_b0 + m * DD, bufC, DD, DD, 1, 1);
        gemm_kernel<<<dim3(NN / 128, DD / 128), 256, 0, stream>>>(
            bufC, mlp1T + (size_t)m * DD * DD, mlp_b1 + m * DD, bufD, DD, DD, 1, 0);
        dot_kernel<<<NN / 8, 256, 0, stream>>>(bufD, mlp_W2 + (size_t)m * DD, mlp_b2, m,
                                               y2 + (size_t)m * NN);
    }

    // ---- bootstrap mean/std ----
    boot_kernel<<<NN / 256, 256, 0, stream>>>(y2, (float*)d_out);
}
